// Net_24764781429129
// MI455X (gfx1250) — matrix operand order repaired and checked by reading, not compiled, not run on hardware
//
#include <hip/hip_runtime.h>

// ---------------- problem constants (match reference) ----------------
#define T_STEPS 201
#define B_TOT   2048
#define NIN     784
#define NH      128
#define NOUT    10
#define BETA    0.9f
#define THR     1.0f

// ---------------- tiling ----------------
#define BLOCK           128          // 4 waves (wave32)
#define WAVES           4
#define ROWS_PER_WAVE   16
#define ROWS_PER_BLOCK  (WAVES * ROWS_PER_WAVE)   // 64
#define GRID            (B_TOT / ROWS_PER_BLOCK)  // 32

// LDS layout (in _Float16 elements). Row strides padded for bank-conflict-free
// 16-lane strided access (stride must not be 0 mod 64 dwords).
#define W1_KP     808                               // 784 data + 16 zero-pad + 8 bank pad
#define SPK_KP    136                               // 128 data + 8 bank pad
#define W1_H      (NH * W1_KP)                      // 103424 halves
#define SPK_BASE  W1_H
#define SPK_H_PER_WAVE (16 * SPK_KP)                // 2176 halves
#define W2_BASE   (SPK_BASE + WAVES * SPK_H_PER_WAVE)
#define LDS_HALVES (W2_BASE + 16 * SPK_KP)
#define LDS_BYTES  ((size_t)LDS_HALVES * 2)         // 228,608 B  (< 320 KB WGP LDS)

typedef __attribute__((ext_vector_type(16))) _Float16 v16h;
typedef __attribute__((ext_vector_type(8)))  _Float16 v8h;
typedef __attribute__((ext_vector_type(8)))  float    v8f;

union HF16 {
    v16h     v;
    v8h      h[2];
    _Float16 e[16];
};

__global__ __launch_bounds__(BLOCK) void snn_lif_wmma_kernel(
    const float* __restrict__ x,    // [B, T, 784]
    const float* __restrict__ W1,   // [128, 784]
    const float* __restrict__ b1,   // [128]
    const float* __restrict__ W2,   // [10, 128]
    const float* __restrict__ b2,   // [10]
    float* __restrict__ out_spk,    // [T, B, 10]
    float* __restrict__ out_mem)    // [T, B, 10]
{
    extern __shared__ _Float16 smem[];

    const int tid  = threadIdx.x;
    const int lane = tid & 31;
    const int wave = tid >> 5;
    const int lhi  = (lane >> 4) & 1;   // 0: lanes 0-15, 1: lanes 16-31
    const int l16  = lane & 15;

    // ---- stage W1 [128][784] f32 -> LDS f16 [128][W1_KP], one row per thread
    {
        const int n = tid;                      // BLOCK == NH == 128
        const float* src = W1 + (size_t)n * NIN;
        _Float16* dst = smem + (size_t)n * W1_KP;
        for (int k = 0; k < NIN; k += 8) {
            float4 a = *(const float4*)(src + k);
            float4 b = *(const float4*)(src + k + 4);
            v8h h;
            h[0] = (_Float16)a.x; h[1] = (_Float16)a.y;
            h[2] = (_Float16)a.z; h[3] = (_Float16)a.w;
            h[4] = (_Float16)b.x; h[5] = (_Float16)b.y;
            h[6] = (_Float16)b.z; h[7] = (_Float16)b.w;
            *(v8h*)(dst + k) = h;
        }
        v8h z = (v8h)(_Float16)0.0f;            // zero K pad 784..799 (+ bank pad)
        *(v8h*)(dst + 784) = z;
        *(v8h*)(dst + 792) = z;
        *(v8h*)(dst + 800) = z;
    }
    // ---- stage W2 [10][128] f32 -> LDS f16 [16][SPK_KP], zero rows 10..15
    {
        const int h = tid;                      // column 0..127
        #pragma unroll
        for (int o = 0; o < 16; ++o) {
            float v = (o < NOUT) ? W2[(size_t)o * NH + h] : 0.0f;
            smem[W2_BASE + o * SPK_KP + h] = (_Float16)v;
        }
    }
    __syncthreads();

    // ---- per-lane biases (accumulator layout: N = lane&15 within each N-tile)
    float b1l[8];
    #pragma unroll
    for (int nt = 0; nt < 8; ++nt) b1l[nt] = b1[nt * 16 + l16];
    const float b2l = (l16 < NOUT) ? b2[l16] : 0.0f;

    // ---- recurrent state in registers
    v8f mem1[8];
    #pragma unroll
    for (int nt = 0; nt < 8; ++nt) mem1[nt] = (v8f)0.0f;
    v8f mem2 = (v8f)0.0f;

    const int row0 = blockIdx.x * ROWS_PER_BLOCK + wave * ROWS_PER_WAVE;
    const int arow = row0 + l16;                // lanes L and L+16 share a row (A layout)
    const float* xrow_base = x + (size_t)arow * ((size_t)T_STEPS * NIN);

    _Float16* spkbuf = smem + SPK_BASE + wave * SPK_H_PER_WAVE;   // wave-private
    const _Float16* w2l = smem + W2_BASE;

    for (int t = 0; t < T_STEPS; ++t) {
        const float* xr  = xrow_base + (size_t)t * NIN;
        const float* xrn = xrow_base + (size_t)((t + 1 < T_STEPS) ? t + 1 : t) * NIN;

        v8f acc[8];
        #pragma unroll
        for (int nt = 0; nt < 8; ++nt) acc[nt] = (v8f)0.0f;

        // ---------- layer 1: cur1 = x_t @ W1^T, K = 800 (25 steps of 32) ----------
        for (int ks = 0; ks < 25; ++ks) {
            const int k0 = ks * 32 + (lhi ? 8 : 0);     // A layout K base for this lane
            HF16 afr;
            {
                float4 c0 = *(const float4*)(xr + k0);
                float4 c1 = *(const float4*)(xr + k0 + 4);
                afr.e[0] = (_Float16)c0.x; afr.e[1] = (_Float16)c0.y;
                afr.e[2] = (_Float16)c0.z; afr.e[3] = (_Float16)c0.w;
                afr.e[4] = (_Float16)c1.x; afr.e[5] = (_Float16)c1.y;
                afr.e[6] = (_Float16)c1.z; afr.e[7] = (_Float16)c1.w;
                if (ks < 24) {                           // K tail 784..799 is zero
                    float4 c2 = *(const float4*)(xr + k0 + 16);
                    float4 c3 = *(const float4*)(xr + k0 + 20);
                    afr.e[ 8] = (_Float16)c2.x; afr.e[ 9] = (_Float16)c2.y;
                    afr.e[10] = (_Float16)c2.z; afr.e[11] = (_Float16)c2.w;
                    afr.e[12] = (_Float16)c3.x; afr.e[13] = (_Float16)c3.y;
                    afr.e[14] = (_Float16)c3.z; afr.e[15] = (_Float16)c3.w;
                } else {
                    afr.h[1] = (v8h)(_Float16)0.0f;
                }
            }
            __builtin_prefetch(xrn + k0, 0, 3);          // global_prefetch_b8: next timestep

            const int kkb = ks * 32 + (lhi ? 8 : 0);     // B layout K base for this lane
            #pragma unroll
            for (int nt = 0; nt < 8; ++nt) {
                HF16 bfr;
                const _Float16* bp = smem + (size_t)(nt * 16 + l16) * W1_KP + kkb;
                bfr.h[0] = *(const v8h*)bp;
                bfr.h[1] = *(const v8h*)(bp + 16);
                acc[nt] = __builtin_amdgcn_wmma_f32_16x16x32_f16(
                    false, afr.v, false, bfr.v, (short)0, acc[nt], false, false);
            }
        }

        // ---------- LIF layer 1 + spike spill to LDS (A-friendly layout) ----------
        #pragma unroll
        for (int nt = 0; nt < 8; ++nt) {
            const int ncol = nt * 16 + l16;
            #pragma unroll
            for (int r = 0; r < 8; ++r) {
                float cur = acc[nt][r] + b1l[nt];
                float old = mem1[nt][r];
                float rst = (old > THR) ? THR : 0.0f;    // detached reset
                float m   = BETA * old + cur - rst;
                mem1[nt][r] = m;
                _Float16 s = (m > THR) ? (_Float16)1.0f : (_Float16)0.0f;
                const int mrow = r + (lhi ? 8 : 0);      // C/D layout row for this lane
                spkbuf[mrow * SPK_KP + ncol] = s;
            }
        }
        __syncthreads();

        // ---------- layer 2: cur2 = spk1 @ W2^T, K = 128 (4 steps of 32) ----------
        v8f acc2 = (v8f)0.0f;
        #pragma unroll
        for (int ks2 = 0; ks2 < 4; ++ks2) {
            HF16 afr, bfr;
            const _Float16* ap = spkbuf + l16 * SPK_KP + ks2 * 32 + (lhi ? 8 : 0);
            afr.h[0] = *(const v8h*)ap;
            afr.h[1] = *(const v8h*)(ap + 16);
            const _Float16* bp = w2l + l16 * SPK_KP + ks2 * 32 + (lhi ? 8 : 0);
            bfr.h[0] = *(const v8h*)bp;
            bfr.h[1] = *(const v8h*)(bp + 16);
            acc2 = __builtin_amdgcn_wmma_f32_16x16x32_f16(
                false, afr.v, false, bfr.v, (short)0, acc2, false, false);
        }

        // ---------- LIF layer 2 + outputs ----------
        const int o = l16;
        #pragma unroll
        for (int r = 0; r < 8; ++r) {
            float cur = acc2[r] + b2l;
            float old = mem2[r];
            float rst = (old > THR) ? THR : 0.0f;
            float m   = BETA * old + cur - rst;
            mem2[r]   = m;
            float s   = (m > THR) ? 1.0f : 0.0f;
            if (o < NOUT) {
                const int grow = row0 + r + (lhi ? 8 : 0);
                const size_t oi = ((size_t)t * B_TOT + (size_t)grow) * NOUT + o;
                out_spk[oi] = s;
                out_mem[oi] = m;
            }
        }
    }
}

extern "C" void kernel_launch(void* const* d_in, const int* in_sizes, int n_in,
                              void* d_out, int out_size, void* d_ws, size_t ws_size,
                              hipStream_t stream) {
    (void)in_sizes; (void)n_in; (void)d_ws; (void)ws_size; (void)out_size;
    const float* x  = (const float*)d_in[0];
    const float* W1 = (const float*)d_in[1];
    const float* b1 = (const float*)d_in[2];
    const float* W2 = (const float*)d_in[3];
    const float* b2 = (const float*)d_in[4];
    float* out_spk = (float*)d_out;                                  // [201,2048,10]
    float* out_mem = out_spk + (size_t)T_STEPS * B_TOT * NOUT;       // [201,2048,10]

    snn_lif_wmma_kernel<<<GRID, BLOCK, LDS_BYTES, stream>>>(
        x, W1, b1, W2, b2, out_spk, out_mem);
}
